// GRU_51376398794979
// MI455X (gfx1250) — matrix operand order repaired and checked by reading, not compiled, not run on hardware
//
#include <hip/hip_runtime.h>
#include <math.h>

// ---------------------------------------------------------------------------
// Problem constants (match reference)
// ---------------------------------------------------------------------------
#define B_   128
#define S_   512
#define D_   512
#define T_   64
#define OUT_ 3
#define KC   64   // K-chunk per LDS stage (two 16x16x32 WMMAs)

typedef __attribute__((ext_vector_type(16))) __bf16 v16bf;
typedef __attribute__((ext_vector_type(8)))  float  v8f;

// float -> bf16 bits, round-to-nearest-even
static __device__ __forceinline__ unsigned f2bf(float f) {
    union { float f; unsigned u; } v; v.f = f;
    unsigned r = v.u + 0x7FFFu + ((v.u >> 16) & 1u);
    return r >> 16;
}

union FragCast {
    uint4 q[2];
    v16bf v;
};

// ---------------------------------------------------------------------------
// One-time weight repack: W[N, ldw] fp32 (first K cols) -> Wb[N, K] bf16
// one thread per 4 elements along k
// ---------------------------------------------------------------------------
__global__ __launch_bounds__(256)
void pack_w_bf16(const float* __restrict__ W, int ldw,
                 unsigned short* __restrict__ Wb, int K) {
    const int i  = blockIdx.x * blockDim.x + threadIdx.x;  // N*K/4 threads
    const int n  = i / (K / 4);
    const int k4 = (i % (K / 4)) * 4;
    const float* src = W + (size_t)n * ldw + k4;
    uint2 p;
    p.x = f2bf(src[0]) | (f2bf(src[1]) << 16);
    p.y = f2bf(src[2]) | (f2bf(src[3]) << 16);
    *(uint2*)&Wb[(size_t)n * K + k4] = p;
}

// ---------------------------------------------------------------------------
// C[M,N] = A[M,K] @ Wb[N,K]^T + bias[N]
// A fp32, Wb bf16 (k-contiguous, stride K), C fp32. bf16 WMMA core.
// Block = 256 threads (8 waves). Block tile: 16 (M) x 128 (N).
// grid.x = N/128, grid.y = M/16. K multiple of 64.
//
// LDS layouts are fragment-native: each lane's WMMA fragment is two
// contiguous 16B LDS reads (ds_load_b128).
// ---------------------------------------------------------------------------
__global__ __launch_bounds__(256)
void gemm_xwt_wmma(const float* __restrict__ A, int lda,
                   const unsigned short* __restrict__ Wb,
                   const float* __restrict__ bias,
                   float* __restrict__ C, int ldc, int K) {
    __shared__ __align__(16) unsigned short As[16][KC];    // [m][k]  2 KB
    __shared__ __align__(16) unsigned short Bs[128][KC];   // [n][k] 16 KB

    const int tid   = threadIdx.x;
    const int lane  = tid & 31;
    const int wave  = tid >> 5;
    const int mbase = blockIdx.y * 16;
    const int nblk  = blockIdx.x * 128;
    const int wn    = wave * 16;               // wave's N offset in block tile

    const int mfrag = lane & 15;               // A row this lane holds
    const int nfrag = wn + (lane & 15);        // B column this lane holds
    const int kgA   = (lane & 16) ? 8 : 0;     // A K-group (ISA 16-bit A layout)
    const int kgB   = (lane & 16) ? 8 : 0;     // B K-group (same as A K-group)

    // fixed staging assignments (exact, no loops)
    const int am  = tid >> 4;                  // A: row 0..15
    const int ak4 = (tid & 15) << 2;           // A: k offset 0..60 (float4)

    v8f acc = {};

    for (int kk = 0; kk < K; kk += KC) {
        // ---- stage A tile: one float4 -> 4 bf16 -> ds_store_b64 per thread
        {
            const float4 av = *(const float4*)&A[(size_t)(mbase + am) * lda + kk + ak4];
            uint2 p;
            p.x = f2bf(av.x) | (f2bf(av.y) << 16);
            p.y = f2bf(av.z) | (f2bf(av.w) << 16);
            *(uint2*)&As[am][ak4] = p;
        }
        // ---- stage B tile: 4x uint4 (8 bf16 each) straight copies per thread
#pragma unroll
        for (int j = 0; j < 4; ++j) {
            const int flat = j * 256 + tid;    // 1024 x 16B chunks
            const int n  = flat >> 3;          // 0..127
            const int kc = (flat & 7) << 3;    // ushort offset 0..56
            *(uint4*)&Bs[n][kc] =
                *(const uint4*)&Wb[(size_t)(nblk + n) * K + kk + kc];
        }
        // prefetch next K-chunk of weights (global_prefetch_b8)
        if (kk + KC < K) {
            __builtin_prefetch(&Wb[(size_t)(nblk + (tid >> 1)) * K + kk + KC +
                                   (tid & 1) * 32], 0, 2);
        }
        __syncthreads();

#pragma unroll
        for (int half = 0; half < 2; ++half) {
            const int ko = half * 32;
            // A fragment: a[0..7] = As[m][kgA..kgA+7], a[8..15] = +16
            FragCast af;
            af.q[0] = *(const uint4*)&As[mfrag][ko + kgA];
            af.q[1] = *(const uint4*)&As[mfrag][ko + kgA + 16];
            // B fragment: b[0..7] = Bs[n][kgB..kgB+7], b[8..15] = +16
            FragCast bf;
            bf.q[0] = *(const uint4*)&Bs[nfrag][ko + kgB];
            bf.q[1] = *(const uint4*)&Bs[nfrag][ko + kgB + 16];

            acc = __builtin_amdgcn_wmma_f32_16x16x32_bf16(
                /*neg_a=*/false, af.v, /*neg_b=*/false, bf.v,
                /*c_mod=*/(short)0, acc, /*reuse_a=*/false, /*reuse_b=*/false);
        }
        __syncthreads();
    }

    // epilogue: C layout -> VGPR r: M = r + (lane>=16 ? 8 : 0), N = lane&15
    const int nglob = nblk + wn + (lane & 15);
    const float bv  = bias ? bias[nglob] : 0.0f;
    const int mhi   = (lane & 16) ? 8 : 0;
#pragma unroll
    for (int r = 0; r < 8; ++r) {
        int m = mbase + r + mhi;
        C[(size_t)m * ldc + nglob] = acc[r] + bv;
    }
}

// ---------------------------------------------------------------------------
// scores[b,s] = sum_d va[d] * tanh(q[b,d] + Uk[b,s,d]) + vab
// One wave per (b,s), float4-vectorized. Block = 256 (8 waves).
// ---------------------------------------------------------------------------
__global__ __launch_bounds__(256)
void scores_kernel(const float* __restrict__ q, const float* __restrict__ Uk,
                   const float* __restrict__ va, const float* __restrict__ vab,
                   float* __restrict__ sc) {
    const int p    = blockIdx.x * 8 + (threadIdx.x >> 5);
    const int lane = threadIdx.x & 31;
    const int b = p >> 9, s = p & (S_ - 1);
    const float4* q4 = (const float4*)(q + (size_t)b * D_);
    const float4* u4 = (const float4*)(Uk + ((size_t)b * S_ + s) * D_);
    const float4* v4 = (const float4*)va;
    float acc = 0.0f;
#pragma unroll
    for (int i = 0; i < D_ / 128; ++i) {
        const int idx = i * 32 + lane;
        const float4 qv = q4[idx], uv = u4[idx], vv = v4[idx];
        acc += vv.x * tanhf(qv.x + uv.x);
        acc += vv.y * tanhf(qv.y + uv.y);
        acc += vv.z * tanhf(qv.z + uv.z);
        acc += vv.w * tanhf(qv.w + uv.w);
    }
#pragma unroll
    for (int off = 16; off > 0; off >>= 1)
        acc += __shfl_xor(acc, off);
    if (lane == 0) sc[(size_t)b * S_ + s] = acc + vab[0];
}

// ---------------------------------------------------------------------------
// row softmax over S=512; writes directly into cross_attn[b,t,:] slice
// ---------------------------------------------------------------------------
__global__ __launch_bounds__(256)
void softmax_kernel(const float* __restrict__ sc, float* __restrict__ attn, int t) {
    __shared__ float red[256];
    const int b   = blockIdx.x;
    const int tid = threadIdx.x;
    const float* row = sc + (size_t)b * S_;
    float* out = attn + ((size_t)b * T_ + t) * S_;

    float v0 = row[tid], v1 = row[tid + 256];
    red[tid] = fmaxf(v0, v1);
    __syncthreads();
    for (int off = 128; off > 0; off >>= 1) {
        if (tid < off) red[tid] = fmaxf(red[tid], red[tid + off]);
        __syncthreads();
    }
    const float mx = red[0];
    __syncthreads();
    float e0 = __expf(v0 - mx), e1 = __expf(v1 - mx);
    red[tid] = e0 + e1;
    __syncthreads();
    for (int off = 128; off > 0; off >>= 1) {
        if (tid < off) red[tid] += red[tid + off];
        __syncthreads();
    }
    const float inv = 1.0f / red[0];
    out[tid]       = e0 * inv;
    out[tid + 256] = e1 * inv;
}

// ---------------------------------------------------------------------------
// ctx[b,d] = sum_s w[b,s] * e_all[b,s,d]; one thread per (b, 4d), float4 loads
// ---------------------------------------------------------------------------
__global__ __launch_bounds__(256)
void ctx_kernel(const float* __restrict__ attn, const float* __restrict__ e_all,
                float* __restrict__ ctx, int t) {
    const int idx = blockIdx.x * blockDim.x + threadIdx.x; // B*D/4 threads
    const int b = idx >> 7, d4 = idx & (D_ / 4 - 1);
    const float* wr = attn + ((size_t)b * T_ + t) * S_;
    const float4* er = (const float4*)(e_all + (size_t)b * S_ * D_) + d4;
    float4 acc = {0.f, 0.f, 0.f, 0.f};
#pragma unroll 4
    for (int s = 0; s < S_; ++s) {
        const float w = wr[s];
        const float4 e = er[(size_t)s * (D_ / 4)];
        acc.x += w * e.x; acc.y += w * e.y;
        acc.z += w * e.z; acc.w += w * e.w;
    }
    ((float4*)ctx)[idx] = acc;
}

// ---------------------------------------------------------------------------
// GRU pointwise: folds in the x-part of gi (W_ih columns 512..514)
// ---------------------------------------------------------------------------
__global__ __launch_bounds__(256)
void gru_pointwise(const float* __restrict__ gi, const float* __restrict__ gh,
                   const float* __restrict__ hin, const float* __restrict__ x,
                   int xstride, const float* __restrict__ W_ih,
                   float* __restrict__ hout) {
    const int idx = blockIdx.x * blockDim.x + threadIdx.x; // B*D
    const int b = idx >> 9, d = idx & (D_ - 1);
    float c0 = 0.f, c1 = 0.f, c2 = 0.f;
    if (x) {
        const float x0 = x[(size_t)b * xstride + 0];
        const float x1 = x[(size_t)b * xstride + 1];
        const float x2 = x[(size_t)b * xstride + 2];
        const float* w0 = W_ih + (size_t)(d)          * (D_ + OUT_) + D_;
        const float* w1 = W_ih + (size_t)(D_ + d)     * (D_ + OUT_) + D_;
        const float* w2 = W_ih + (size_t)(2 * D_ + d) * (D_ + OUT_) + D_;
        c0 = x0 * w0[0] + x1 * w0[1] + x2 * w0[2];
        c1 = x0 * w1[0] + x1 * w1[1] + x2 * w1[2];
        c2 = x0 * w2[0] + x1 * w2[1] + x2 * w2[2];
    }
    const int base = b * (3 * D_) + d;
    const float ir = gi[base] + c0, iz = gi[base + D_] + c1, in = gi[base + 2 * D_] + c2;
    const float hr = gh[base],      hz = gh[base + D_],      hn = gh[base + 2 * D_];
    const float r = 1.0f / (1.0f + __expf(-(ir + hr)));
    const float z = 1.0f / (1.0f + __expf(-(iz + hz)));
    const float n = tanhf(in + r * hn);
    hout[idx] = (1.0f - z) * n + z * hin[idx];
}

// ---------------------------------------------------------------------------
// out[b,t,j] = h[b,:] . W_out[j,:] + b_out[j]; one wave per (b,j)
// ---------------------------------------------------------------------------
__global__ __launch_bounds__(256)
void outproj_kernel(const float* __restrict__ h, const float* __restrict__ W_out,
                    const float* __restrict__ b_out, float* __restrict__ dout, int t) {
    const int p    = blockIdx.x * 8 + (threadIdx.x >> 5);
    const int lane = threadIdx.x & 31;
    if (p >= B_ * OUT_) return;
    const int b = p / OUT_, j = p % OUT_;
    float acc = 0.0f;
#pragma unroll 4
    for (int d = lane; d < D_; d += 32)
        acc += h[(size_t)b * D_ + d] * W_out[(size_t)j * D_ + d];
#pragma unroll
    for (int off = 16; off > 0; off >>= 1)
        acc += __shfl_xor(acc, off);
    if (lane == 0) dout[((size_t)b * T_ + t) * OUT_ + j] = acc + b_out[j];
}

// ---------------------------------------------------------------------------
// host side
// ---------------------------------------------------------------------------
extern "C" void kernel_launch(void* const* d_in, const int* in_sizes, int n_in,
                              void* d_out, int out_size, void* d_ws, size_t ws_size,
                              hipStream_t stream) {
    (void)in_sizes; (void)n_in; (void)out_size; (void)ws_size;

    const float* e_all  = (const float*)d_in[0];
    const float* e_last = (const float*)d_in[1];
    const float* target = (const float*)d_in[2];
    const float* Wa     = (const float*)d_in[3];
    const float* ba     = (const float*)d_in[4];
    const float* Ua     = (const float*)d_in[5];
    const float* bu     = (const float*)d_in[6];
    const float* Va_w   = (const float*)d_in[7];
    const float* Va_b   = (const float*)d_in[8];
    const float* W_ih   = (const float*)d_in[9];
    const float* b_ih   = (const float*)d_in[10];
    const float* W_hh   = (const float*)d_in[11];
    const float* b_hh   = (const float*)d_in[12];
    const float* W_out  = (const float*)d_in[13];
    const float* b_out  = (const float*)d_in[14];

    float* outD    = (float*)d_out;                 // [B,T,3]
    float* outH    = outD + (size_t)B_ * T_ * OUT_; // [1,B,D]
    float* outAttn = outH + (size_t)B_ * D_;        // [B,T,S]

    // workspace layout (fp32 part)
    float* Uk  = (float*)d_ws;                       // [B,S,D] = 128 MB
    float* q   = Uk  + (size_t)B_ * S_ * D_;
    float* sc  = q   + (size_t)B_ * D_;
    float* ctx = sc  + (size_t)B_ * S_;
    float* gi  = ctx + (size_t)B_ * D_;
    float* gh  = gi  + (size_t)B_ * 3 * D_;
    float* h0  = gh  + (size_t)B_ * 3 * D_;
    float* h1  = h0  + (size_t)B_ * D_;
    // bf16 packed weights
    unsigned short* Ua_b  = (unsigned short*)(h1 + (size_t)B_ * D_);
    unsigned short* Wa_b  = Ua_b  + (size_t)D_ * D_;
    unsigned short* Wih_b = Wa_b  + (size_t)D_ * D_;
    unsigned short* Whh_b = Wih_b + (size_t)3 * D_ * D_;

    // h <- e_last[0]
    hipMemcpyAsync(h0, e_last, (size_t)B_ * D_ * sizeof(float),
                   hipMemcpyDeviceToDevice, stream);

    // one-time bf16 weight packs
    pack_w_bf16<<<(D_ * D_ / 4) / 256, 256, 0, stream>>>(Ua, D_, Ua_b, D_);
    pack_w_bf16<<<(D_ * D_ / 4) / 256, 256, 0, stream>>>(Wa, D_, Wa_b, D_);
    pack_w_bf16<<<(3 * D_ * D_ / 4) / 256, 256, 0, stream>>>(W_ih, D_ + OUT_, Wih_b, D_);
    pack_w_bf16<<<(3 * D_ * D_ / 4) / 256, 256, 0, stream>>>(W_hh, D_, Whh_b, D_);

    // Uk = e_all @ Ua^T + bu   (65536 x 512 x 512, bf16 WMMA)
    gemm_xwt_wmma<<<dim3(D_ / 128, (B_ * S_) / 16), 256, 0, stream>>>(
        e_all, D_, Ua_b, bu, Uk, D_, D_);

    for (int t = 0; t < T_; ++t) {
        float* hin  = (t & 1) ? h1 : h0;
        float* hout = (t & 1) ? h0 : h1;

        // q = h @ Wa^T + ba
        gemm_xwt_wmma<<<dim3(D_ / 128, B_ / 16), 256, 0, stream>>>(
            hin, D_, Wa_b, ba, q, D_, D_);

        // scores / softmax / ctx
        scores_kernel<<<(B_ * S_) / 8, 256, 0, stream>>>(q, Uk, Va_w, Va_b, sc);
        softmax_kernel<<<B_, 256, 0, stream>>>(sc, outAttn, t);
        ctx_kernel<<<(B_ * D_ / 4) / 256, 256, 0, stream>>>(outAttn, e_all, ctx, t);

        // gi = ctx @ W_ih[:, :512]^T + b_ih  (x-part folded into pointwise)
        gemm_xwt_wmma<<<dim3((3 * D_) / 128, B_ / 16), 256, 0, stream>>>(
            ctx, D_, Wih_b, b_ih, gi, 3 * D_, D_);
        // gh = h @ W_hh^T + b_hh
        gemm_xwt_wmma<<<dim3((3 * D_) / 128, B_ / 16), 256, 0, stream>>>(
            hin, D_, Whh_b, b_hh, gh, 3 * D_, D_);

        const float* xptr = (t == 0) ? nullptr : (target + (size_t)(t - 1) * OUT_);
        gru_pointwise<<<(B_ * D_) / 256, 256, 0, stream>>>(
            gi, gh, hin, xptr, T_ * OUT_, W_ih, hout);

        outproj_kernel<<<(B_ * OUT_ + 7) / 8, 256, 0, stream>>>(
            hout, W_out, b_out, outD, t);
    }

    // final hidden ends in h0 after 64 steps (even count)
    hipMemcpyAsync(outH, h0, (size_t)B_ * D_ * sizeof(float),
                   hipMemcpyDeviceToDevice, stream);
}
